// MemoryGraph_85237920956707
// MI455X (gfx1250) — matrix operand order repaired and checked by reading, not compiled, not run on hardware
//
#include <hip/hip_runtime.h>
#include <hip/hip_bf16.h>

typedef __attribute__((ext_vector_type(16))) __bf16 v16bf;
typedef __attribute__((ext_vector_type(8)))  float  v8f;
typedef __attribute__((ext_vector_type(8)))  unsigned short u16x8;
typedef __attribute__((ext_vector_type(4)))  unsigned int u32x4;
typedef __attribute__((ext_vector_type(8)))  int i32x8;
typedef __attribute__((ext_vector_type(4)))  int i32x4;

#define B_     8
#define NC_    64
#define N_     256
#define DN_    64
#define ALPHA_ 8
#define SIN_   192
#define BT_LD  264   // padded ld for transposed B tile in k_agg

// ---------- helpers ----------
__device__ __forceinline__ unsigned short f2bfu(float f) {
  union { float f; unsigned u; } v; v.f = f;
  unsigned r = (v.u + 0x7FFFu + ((v.u >> 16) & 1u)) >> 16;
  return (unsigned short)r;
}
// packed f32x2 -> bf16x2 (RNE) via CDNA5 V_CVT_PK_BF16_F32
__device__ __forceinline__ unsigned pk_bf16(float lo, float hi) {
  unsigned r;
  asm("v_cvt_pk_bf16_f32 %0, %1, %2" : "=v"(r) : "v"(lo), "v"(hi));
  return r;
}
__device__ __forceinline__ v8f wmma_bf16(v16bf a, v16bf b, v8f c) {
  return __builtin_amdgcn_wmma_f32_16x16x32_bf16(false, a, false, b, (short)0, c, false, false);
}

// A fragment 16x32 bf16 from row-major [M][K] src, two contiguous 8-elem runs per lane.
__device__ __forceinline__ v16bf load_a_frag(const unsigned short* src, int ld,
                                             int rbase, int kbase, int lane) {
  const unsigned short* row = src + (size_t)(rbase + (lane & 15)) * ld + kbase + ((lane >> 4) * 8);
  union { u16x8 u[2]; v16bf v; } c;
  c.u[0] = *(const u16x8*)(row);
  c.u[1] = *(const u16x8*)(row + 16);
  return c.v;
}
// B fragment 32x16 bf16 from [N][K] (k-contiguous) src, two contiguous 8-elem runs per lane.
__device__ __forceinline__ v16bf load_b_frag_nk(const unsigned short* src, int ld,
                                                int nbase, int kbase, int lane) {
  const unsigned short* p = src + (size_t)(nbase + (lane & 15)) * ld + kbase + ((lane >> 4) * 8);
  union { u16x8 u[2]; v16bf v; } c;
  c.u[0] = *(const u16x8*)(p);
  c.u[1] = *(const u16x8*)(p + 16);
  return c.v;
}

// ---------- CDNA5 async global->LDS (ASYNCcnt) ----------
__device__ __forceinline__ unsigned lds_off(const void* p) {
  return (unsigned)(unsigned long long)p;   // addr[31:0] = LDS byte offset
}
__device__ __forceinline__ void async_ld_b128(unsigned ldsoff, const void* g) {
  asm volatile("global_load_async_to_lds_b128 %0, %1, off"
               :: "v"(ldsoff), "v"((unsigned long long)g) : "memory");
}
__device__ __forceinline__ void wait_asynccnt0() {
  asm volatile("s_wait_asynccnt 0x0" ::: "memory");
}

// ---------- CDNA5 TDM: contiguous nbytes (multiple of 8) global -> LDS ----------
__device__ __forceinline__ void tdm_load_1d(unsigned ldsoff, const void* g, unsigned n8) {
#if __has_builtin(__builtin_amdgcn_tensor_load_to_lds)
  unsigned long long ga = (unsigned long long)g;
  u32x4 g0;
  g0[0] = 1u;                                   // count=1, user descriptor
  g0[1] = ldsoff;                               // lds_addr
  g0[2] = (unsigned)ga;                         // global_addr[31:0] -> bits 95:64
  g0[3] = (unsigned)((ga >> 32) & 0x01FFFFFFu)  // global_addr[56:32]
        | (2u << 30);                           // type = 2 ("image")
  i32x8 g1;
  g1[0] = (3 << 16);                            // data_size = 8B; mask/flags = 0
  g1[1] = (int)(n8 << 16);                      // tensor_dim0 low16 (bits 63:48)
  g1[2] = (int)(n8 >> 16);                      // tensor_dim0 high16 (bits 79:64)
  g1[3] = (int)(n8 << 16);                      // tile_dim0 (bits 127:112)
  g1[4] = 0;                                    // tile_dim1/2 unused
  g1[5] = (int)n8;                              // tensor_dim0_stride low32
  g1[6] = 0;
  g1[7] = 0;
  i32x4 z4 = {0, 0, 0, 0};
  i32x8 z8 = {0, 0, 0, 0, 0, 0, 0, 0};
  __builtin_amdgcn_tensor_load_to_lds(g0, g1, z4, z4, z8, 0);
  __builtin_amdgcn_s_wait_tensorcnt(0);
#else
  // fallback: per-lane async copies (32 lanes assumed active)
  int lane = threadIdx.x & 31;
  for (unsigned b = lane * 16; b < n8 * 8; b += 32 * 16)
    async_ld_b128(ldsoff + b, (const char*)g + b);
  wait_asynccnt0();
#endif
}

// ---------- trivial kernels ----------
__global__ void k_zero(float* p, int n) {
  int i = blockIdx.x * blockDim.x + threadIdx.x;
  if (i < n) p[i] = 0.f;
}
// pairwise f32 -> packed bf16 convert
__global__ void k_cvt2(const float* __restrict__ src, unsigned* __restrict__ dst, int npair) {
  int i = blockIdx.x * blockDim.x + threadIdx.x;
  int st = gridDim.x * blockDim.x;
  for (; i < npair; i += st) dst[i] = pk_bf16(src[2 * i], src[2 * i + 1]);
}

// ---------- inject ----------
__global__ void k_inject(const float* __restrict__ x, const float* __restrict__ h,
                         const float* __restrict__ iw, const float* __restrict__ ib,
                         float* __restrict__ hinjF, unsigned short* __restrict__ hinjB) {
  int cell = blockIdx.x;
  int c = cell & (NC_ - 1);
  int t = threadIdx.x;
  __shared__ float xs[DN_];
  __shared__ float injs[ALPHA_ * DN_];
  if (t < DN_) xs[t] = x[cell * DN_ + t];
  __syncthreads();
  for (int e = t; e < ALPHA_ * DN_; e += 256) {
    const float* wr = iw + ((size_t)c * ALPHA_ * DN_ + e) * DN_;
    float s = ib[c * ALPHA_ * DN_ + e];
#pragma unroll 8
    for (int d = 0; d < DN_; ++d) s += xs[d] * wr[d];
    injs[e] = s;
  }
  __syncthreads();
  size_t base = (size_t)cell * N_ * DN_;
  for (int i = t * 2; i < N_ * DN_; i += 512) {
    float v0 = h[base + i], v1 = h[base + i + 1];
    if (i < ALPHA_ * DN_) { v0 += injs[i]; v1 += injs[i + 1]; }
    hinjF[base + i] = v0; hinjF[base + i + 1] = v1;
    *(unsigned*)(hinjB + base + i) = pk_bf16(v0, v1);
  }
}

// ---------- prep: Whb = bf16(W+heb) + per-cell sum(W^2), sum(heb^2) ----------
__global__ void k_prep(const float* __restrict__ W, const float* __restrict__ heb,
                       unsigned short* __restrict__ Whb,
                       float* __restrict__ ssqW, float* __restrict__ ssqH) {
  int cell = blockIdx.x, t = threadIdx.x;
  size_t base = (size_t)cell * N_ * N_;
  float sw = 0.f, sh = 0.f;
  for (int i = t * 2; i < N_ * N_; i += 512) {
    float w0 = W[base + i], w1 = W[base + i + 1];
    float h0 = heb[base + i], h1 = heb[base + i + 1];
    *(unsigned*)(Whb + base + i) = pk_bf16(w0 + h0, w1 + h1);
    sw += w0 * w0 + w1 * w1; sh += h0 * h0 + h1 * h1;
  }
  __shared__ float r1[256], r2[256];
  r1[t] = sw; r2[t] = sh; __syncthreads();
  for (int s = 128; s > 0; s >>= 1) {
    if (t < s) { r1[t] += r1[t + s]; r2[t] += r2[t + s]; }
    __syncthreads();
  }
  if (t == 0) { ssqW[cell] = r1[0]; ssqH[cell] = r2[0]; }
}

// ---------- agg = (W+heb) @ h_inj ----------
__global__ void __launch_bounds__(256) k_agg(const unsigned short* __restrict__ Whb,
                                             const unsigned short* __restrict__ hinjB,
                                             unsigned short* __restrict__ aggB) {
  int cell = blockIdx.x;
  int t = threadIdx.x, w = t >> 5, lane = t & 31;
  __shared__ unsigned short BshT[DN_ * BT_LD];   // h_inj^T : [d][m]
  size_t hbase = (size_t)cell * N_ * DN_;
  const unsigned short* A = Whb + (size_t)cell * N_ * N_;
  for (int i = t; i < N_ * DN_; i += 256) {
    int m = i >> 6, d = i & 63;
    BshT[d * BT_LD + m] = hinjB[hbase + i];
  }
  __syncthreads();
  v8f acc[2][4] = {};
  for (int kk = 0; kk < 8; ++kk) {
    v16bf bfr[4];
#pragma unroll
    for (int ct = 0; ct < 4; ++ct) bfr[ct] = load_b_frag_nk(BshT, BT_LD, ct * 16, kk * 32, lane);
#pragma unroll
    for (int rt = 0; rt < 2; ++rt) {
      v16bf a = load_a_frag(A, N_, (2 * w + rt) * 16, kk * 32, lane);
      if (kk < 7)
        __builtin_prefetch(A + (size_t)((2 * w + rt) * 16 + (lane & 15)) * N_ + (kk + 1) * 32, 0, 3);
#pragma unroll
      for (int ct = 0; ct < 4; ++ct) acc[rt][ct] = wmma_bf16(a, bfr[ct], acc[rt][ct]);
    }
  }
#pragma unroll
  for (int rt = 0; rt < 2; ++rt)
#pragma unroll
    for (int ct = 0; ct < 4; ++ct)
#pragma unroll
      for (int r = 0; r < 8; ++r) {
        int row = (2 * w + rt) * 16 + (lane >> 4) * 8 + r;
        int col = ct * 16 + (lane & 15);
        aggB[hbase + row * DN_ + col] = f2bfu(acc[rt][ct][r]);
      }
}

// ---------- fused 2-layer MLP: 32 rows/block, 192->256 tanh ->64 ----------
__global__ void __launch_bounds__(256) k_mlp(
    const unsigned short* __restrict__ inA, const unsigned short* __restrict__ inB,
    const unsigned short* __restrict__ nidB,
    const unsigned short* __restrict__ w1b, const float* __restrict__ b1,
    const unsigned short* __restrict__ w2b, const float* __restrict__ b2,
    float* __restrict__ outF, unsigned short* __restrict__ outB, int mode,
    const float* __restrict__ hinjF, const float* __restrict__ decay,
    float* __restrict__ readout,
    float* __restrict__ statSum, float* __restrict__ statAbs) {
  int t = threadIdx.x, w = t >> 5, lane = t & 31;
  int R0 = blockIdx.x * 32;
  int cell = R0 >> 8;
  __shared__ unsigned short Xsh[32 * SIN_];   // [row][i]
  __shared__ unsigned short Hsh[32 * 256];    // [row][o]
  __shared__ float red[256];

  // async-pack X = [inA | inB | nid] : 768 16-byte chunks, 3 per thread
  {
    unsigned xoff = lds_off(Xsh);
    for (int q = t; q < 768; q += 256) {
      int row = q / 24, rem = q - row * 24;
      int seg = rem >> 3, c16 = rem & 7;
      int R = R0 + row;
      const unsigned short* g;
      if (seg == 0)      g = inA + (size_t)R * 64 + c16 * 8;
      else if (seg == 1) g = inB + (size_t)R * 64 + c16 * 8;
      else               g = nidB + (size_t)(R & (NC_ * N_ - 1)) * 64 + c16 * 8;
      async_ld_b128(xoff + (unsigned)(row * (SIN_ * 2) + seg * 128 + c16 * 16), g);
    }
    wait_asynccnt0();
  }
  __syncthreads();

  // GEMM1
  v8f acc1[4] = {};
  int rt = w & 1, ctb = (w >> 1) * 4;
  for (int kk = 0; kk < 6; ++kk) {
    v16bf a = load_a_frag(Xsh, SIN_, rt * 16, kk * 32, lane);
#pragma unroll
    for (int j = 0; j < 4; ++j) {
      v16bf b = load_b_frag_nk(w1b, SIN_, (ctb + j) * 16, kk * 32, lane);
      acc1[j] = wmma_bf16(a, b, acc1[j]);
    }
  }
#pragma unroll
  for (int j = 0; j < 4; ++j)
#pragma unroll
    for (int r = 0; r < 8; ++r) {
      int row = rt * 16 + (lane >> 4) * 8 + r;
      int col = (ctb + j) * 16 + (lane & 15);
      Hsh[row * 256 + col] = f2bfu(tanhf(acc1[j][r] + b1[col]));
    }
  __syncthreads();

  // GEMM2
  v8f acc2 = {};
  int rt2 = w & 1, ct2 = w >> 1;
  for (int kk = 0; kk < 8; ++kk) {
    v16bf a = load_a_frag(Hsh, 256, rt2 * 16, kk * 32, lane);
    v16bf b = load_b_frag_nk(w2b, 256, ct2 * 16, kk * 32, lane);
    acc2 = wmma_bf16(a, b, acc2);
  }
  float s = 0.f, sa = 0.f;
#pragma unroll
  for (int r = 0; r < 8; ++r) {
    int row = rt2 * 16 + (lane >> 4) * 8 + r;
    int col = ct2 * 16 + (lane & 15);
    int R = R0 + row;
    size_t oi = (size_t)R * 64 + col;
    float val = acc2[r] + b2[col];
    float outv;
    if (mode == 0) {
      outv = val;
    } else {
      float d = decay[R];
      outv = d * hinjF[oi] + (1.f - d) * tanhf(val);
      int n = R & 255;
      if (n >= ALPHA_ && n < 2 * ALPHA_) {
        int bb = R >> 14, cc = (R >> 8) & 63;
        readout[((size_t)(bb * NC_ + cc) * ALPHA_ + (n - ALPHA_)) * 64 + col] = outv;
      }
    }
    outF[oi] = outv;
    outB[oi] = f2bfu(outv);
    s += outv; sa += fabsf(outv);
  }
  red[t] = s; __syncthreads();
  for (int st = 128; st > 0; st >>= 1) { if (t < st) red[t] += red[t + st]; __syncthreads(); }
  if (t == 0) atomicAdd(statSum + cell, red[0]);
  __syncthreads();
  red[t] = sa; __syncthreads();
  for (int st = 128; st > 0; st >>= 1) { if (t < st) red[t] += red[t + st]; __syncthreads(); }
  if (t == 0) atomicAdd(statAbs + cell, red[0]);
}

// ---------- modulator MLP + gates + decay_new ----------
__global__ void k_mod(const float* __restrict__ decay, const float* __restrict__ x,
                      const float* sumH, const float* sumAbsH,
                      const float* sumM, const float* sumAbsM,
                      const float* ssqW, const float* ssqHeb,
                      const float* mw1, const float* mb1,
                      const float* mw2, const float* mb2,
                      const float* hdl, const float* wdl, const float* dgl,
                      float* __restrict__ hdv, float* __restrict__ wdv,
                      float* __restrict__ decay_out) {
  int cell = blockIdx.x, c = cell & (NC_ - 1), t = threadIdx.x;
  __shared__ float red[256];
  __shared__ float stats[8];
  __shared__ float hm[64];
  __shared__ float gams;
  float dv = decay[cell * N_ + t];
  red[t] = dv; __syncthreads();
  for (int s = 128; s > 0; s >>= 1) { if (t < s) red[t] += red[t + s]; __syncthreads(); }
  if (t == 0) stats[6] = red[0] / (float)N_;
  __syncthreads();
  red[t] = (t < DN_) ? fabsf(x[cell * DN_ + t]) : 0.f; __syncthreads();
  for (int s = 128; s > 0; s >>= 1) { if (t < s) red[t] += red[t + s]; __syncthreads(); }
  if (t == 0) {
    stats[7] = red[0] / (float)DN_;
    stats[0] = sumH[cell]    * (1.f / 16384.f);
    stats[1] = sumAbsH[cell] * (1.f / 16384.f);
    stats[2] = sumM[cell]    * (1.f / 16384.f);
    stats[3] = sumAbsM[cell] * (1.f / 16384.f);
    stats[4] = sqrtf(ssqW[cell]   * (1.f / 65536.f));
    stats[5] = sqrtf(ssqHeb[cell] * (1.f / 65536.f));
  }
  __syncthreads();
  if (t < 64) {
    float a = mb1[c * 64 + t];
#pragma unroll
    for (int i = 0; i < 8; ++i) a += stats[i] * mw1[c * 512 + i * 64 + t];
    hm[t] = tanhf(a);
  }
  __syncthreads();
  if (t < 3) {
    float m = mb2[c * 3 + t];
    for (int j = 0; j < 64; ++j) m += hm[j] * mw2[c * 192 + j * 3 + t];
    if (t == 0)      hdv[cell] = 1.f / (1.f + expf(-(hdl[c] + m)));
    else if (t == 1) wdv[cell] = 1.f / (1.f + expf(-(wdl[c] + m)));
    else             gams      = 1.f / (1.f + expf(-(dgl[c] + m)));
  }
  __syncthreads();
  float g = gams;
  decay_out[cell * N_ + t] = (1.f - g) * dv + g * 0.5f;
}

// ---------- outer-product WMMA + hebbian blend + row rms-norm of W ----------
__global__ void __launch_bounds__(256) k_consolidate(
    const unsigned short* __restrict__ hnewB,
    const float* __restrict__ W_in, const float* __restrict__ heb_in,
    const float* __restrict__ hdv, const float* __restrict__ wdv,
    float* __restrict__ W_out, float* __restrict__ heb_out) {
  int blk = blockIdx.x;
  int cell = blk >> 1, half = blk & 1;
  int t = threadIdx.x, w = t >> 5, lane = t & 31;
  __shared__ unsigned short hsh[N_ * DN_];   // 32KB, row-major [m][d] == [N][K] for B
  size_t hbase = (size_t)cell * N_ * DN_;
  // one TDM descriptor load of the whole 32KB tile, issued by wave 0
  if (t < 32) {
    tdm_load_1d(lds_off(hsh), hnewB + hbase, (N_ * DN_ * 2) / 8);
  }
  __syncthreads();
  v8f acc[16] = {};
  int rbase = half * 128 + w * 16;
#pragma unroll
  for (int kk = 0; kk < 2; ++kk) {
    v16bf a = load_a_frag(hsh, DN_, rbase, kk * 32, lane);
#pragma unroll
    for (int ct = 0; ct < 16; ++ct) {
      v16bf b = load_b_frag_nk(hsh, DN_, ct * 16, kk * 32, lane);
      acc[ct] = wmma_bf16(a, b, acc[ct]);
    }
  }
  float hd = hdv[cell], wd = wdv[cell];
  size_t wbase = (size_t)cell * N_ * N_;
#pragma unroll
  for (int r = 0; r < 8; ++r) {
    int row = rbase + (lane >> 4) * 8 + r;
    float ssq = 0.f;
#pragma unroll
    for (int ct = 0; ct < 16; ++ct) {
      int col = ct * 16 + (lane & 15);
      size_t idx = wbase + (size_t)row * N_ + col;
      float ov = acc[ct][r] * (1.f / (float)DN_);
      float hn = hd * heb_in[idx] + (1.f - hd) * ov;
      heb_out[idx] = hn;
      float tv = wd * W_in[idx] + (1.f - wd) * hn;
      acc[ct][r] = tv;
      ssq += tv * tv;
    }
#pragma unroll
    for (int m = 1; m < 16; m <<= 1) ssq += __shfl_xor(ssq, m, 32);
    float scale = rsqrtf(ssq * (1.f / (float)N_) + 1e-6f);
#pragma unroll
    for (int ct = 0; ct < 16; ++ct) {
      int col = ct * 16 + (lane & 15);
      size_t idx = wbase + (size_t)row * N_ + col;
      W_out[idx] = acc[ct][r] * scale;
    }
  }
}

// ---------- host launcher ----------
extern "C" void kernel_launch(void* const* d_in, const int* in_sizes, int n_in,
                              void* d_out, int out_size, void* d_ws, size_t ws_size,
                              hipStream_t stream) {
  const float* x        = (const float*)d_in[0];
  const float* h        = (const float*)d_in[1];
  const float* msg      = (const float*)d_in[2];
  const float* W        = (const float*)d_in[3];
  const float* heb      = (const float*)d_in[4];
  const float* decay    = (const float*)d_in[5];
  const float* nid      = (const float*)d_in[6];
  const float* state_w1 = (const float*)d_in[7];
  const float* state_b1 = (const float*)d_in[8];
  const float* state_w2 = (const float*)d_in[9];
  const float* state_b2 = (const float*)d_in[10];
  const float* msg_w1   = (const float*)d_in[11];
  const float* msg_b1   = (const float*)d_in[12];
  const float* msg_w2   = (const float*)d_in[13];
  const float* msg_b2   = (const float*)d_in[14];
  const float* inject_w = (const float*)d_in[15];
  const float* inject_b = (const float*)d_in[16];
  const float* mod_w1   = (const float*)d_in[17];
  const float* mod_b1   = (const float*)d_in[18];
  const float* mod_w2   = (const float*)d_in[19];
  const float* mod_b2   = (const float*)d_in[20];
  const float* hdl      = (const float*)d_in[21];
  const float* wdl      = (const float*)d_in[22];
  const float* dgl      = (const float*)d_in[23];

  float* out       = (float*)d_out;
  float* readout   = out;
  float* h_new     = out + 262144;
  float* new_msg   = h_new + 8388608;
  float* W_new     = new_msg + 8388608;
  float* heb_new   = W_new + 33554432;
  float* decay_new = heb_new + 33554432;

  char* ws = (char*)d_ws;
  size_t off = 0;
  float*          hinjF = (float*)(ws + off);          off += 33554432;
  unsigned short* hinjB = (unsigned short*)(ws + off); off += 16777216;
  unsigned short* Whb   = (unsigned short*)(ws + off); off += 67108864;
  unsigned short* aggB  = (unsigned short*)(ws + off); off += 16777216;
  unsigned short* msgB  = (unsigned short*)(ws + off); off += 16777216;
  unsigned short* nmsgB = (unsigned short*)(ws + off); off += 16777216;
  unsigned short* hnewB = (unsigned short*)(ws + off); off += 16777216;
  unsigned short* nidB  = (unsigned short*)(ws + off); off += 2097152;
  unsigned short* w1mB  = (unsigned short*)(ws + off); off += 98304;
  unsigned short* w2mB  = (unsigned short*)(ws + off); off += 32768;
  unsigned short* w1sB  = (unsigned short*)(ws + off); off += 98304;
  unsigned short* w2sB  = (unsigned short*)(ws + off); off += 32768;
  float* statBase = (float*)(ws + off);
  float* sumH    = statBase;        float* sumAbsH = statBase + 512;
  float* sumM    = statBase + 1024; float* sumAbsM = statBase + 1536;
  float* ssqW    = statBase + 2048; float* ssqHeb  = statBase + 2560;
  float* hdv     = statBase + 3072; float* wdv     = statBase + 3584;

  k_zero<<<8, 256, 0, stream>>>(statBase, 2048);
  k_cvt2<<<2048, 256, 0, stream>>>(msg, (unsigned*)msgB, 4194304);
  k_cvt2<<<512, 256, 0, stream>>>(nid, (unsigned*)nidB, 524288);
  k_cvt2<<<24, 256, 0, stream>>>(msg_w1, (unsigned*)w1mB, 24576);
  k_cvt2<<<8, 256, 0, stream>>>(msg_w2, (unsigned*)w2mB, 8192);
  k_cvt2<<<24, 256, 0, stream>>>(state_w1, (unsigned*)w1sB, 24576);
  k_cvt2<<<8, 256, 0, stream>>>(state_w2, (unsigned*)w2sB, 8192);
  k_inject<<<B_ * NC_, 256, 0, stream>>>(x, h, inject_w, inject_b, hinjF, hinjB);
  k_prep<<<B_ * NC_, 256, 0, stream>>>(W, heb, Whb, ssqW, ssqHeb);
  k_agg<<<B_ * NC_, 256, 0, stream>>>(Whb, hinjB, aggB);
  k_mlp<<<B_ * NC_ * 8, 256, 0, stream>>>(aggB, msgB, nidB, w1mB, msg_b1, w2mB, msg_b2,
                                          new_msg, nmsgB, 0, nullptr, nullptr, nullptr,
                                          sumM, sumAbsM);
  k_mlp<<<B_ * NC_ * 8, 256, 0, stream>>>(hinjB, nmsgB, nidB, w1sB, state_b1, w2sB, state_b2,
                                          h_new, hnewB, 1, hinjF, decay, readout,
                                          sumH, sumAbsH);
  k_mod<<<B_ * NC_, 256, 0, stream>>>(decay, x, sumH, sumAbsH, sumM, sumAbsM, ssqW, ssqHeb,
                                      mod_w1, mod_b1, mod_w2, mod_b2, hdl, wdl, dgl,
                                      hdv, wdv, decay_new);
  k_consolidate<<<B_ * NC_ * 2, 256, 0, stream>>>(hnewB, W, heb, hdv, wdv, W_new, heb_new);
}
